// ReprogrammingLayer_34299608826451
// MI455X (gfx1250) — matrix operand order repaired and checked by reading, not compiled, not run on hardware
//
#include <hip/hip_runtime.h>
#include <stdint.h>

// ---------------- CDNA5 WMMA types ----------------
typedef __attribute__((ext_vector_type(16))) __bf16 v16bf;
typedef __attribute__((ext_vector_type(8)))  __bf16 v8bf;
typedef __attribute__((ext_vector_type(8)))  float  v8f;

union Frag16 { v16bf v; v8bf h[2]; };

// A fragment (16x32 bf16): M = lane&15 for both halves; half=lane>>4 selects
// K sub-ranges {0..7,16..23} vs {8..15,24..31}. Two 16B LDS loads.
__device__ __forceinline__ v16bf load_a_frag(const __bf16* row, int half) {
  Frag16 f;
  f.h[0] = *(const v8bf*)(row + 8 * half);
  f.h[1] = *(const v8bf*)(row + 16 + 8 * half);
  return f.v;
}
// B fragment (32x16 bf16) from LDS stored [n][k] (k contiguous):
// n = lane&15, K sub-ranges {0..7,16..23} vs {8..15,24..31} by half. Two 16B LDS loads.
__device__ __forceinline__ v16bf load_b_frag(const __bf16* rowNK, int half) {
  Frag16 f;
  f.h[0] = *(const v8bf*)(rowNK + 8 * half);
  f.h[1] = *(const v8bf*)(rowNK + 16 + 8 * half);
  return f.v;
}

#define WMMA_BF16(af, bf, cc) \
  __builtin_amdgcn_wmma_f32_16x16x32_bf16(false, (af), false, (bf), (short)0, (cc), false, false)

// Async global->LDS copy of one 16B chunk (CDNA5 ASYNCcnt path, no VGPR data).
__device__ __forceinline__ void async_copy_b128(unsigned lds_off, const void* gptr) {
  unsigned long long ga = (unsigned long long)(uintptr_t)gptr;
  asm volatile("global_load_async_to_lds_b128 %0, %1, off"
               :: "v"(lds_off), "v"(ga) : "memory");
}
__device__ __forceinline__ void async_wait0() {
  asm volatile("s_wait_asynccnt 0" ::: "memory");
}
__device__ __forceinline__ unsigned lds_offset(const void* p) {
  return (unsigned)(uintptr_t)p;   // LDS aperture: addr[31:0] is the LDS offset
}

// ---------------- problem constants ----------------
#define B_   16
#define L_   2048
#define BL   32768      // B*L
#define DM   512
#define S_   1000
#define SP   1024       // padded S
#define DL   1024
#define H_   8
#define E_   64
#define HE   512        // H*E

// ============================================================
// GEMM 1: C_bf16[M,N] = A_f32[M,K] @ W_f32[K,N] + bias
// block = 128 thr (4 waves), tile 64x64, K-step 32
// ============================================================
__global__ __launch_bounds__(128)
void reprog_gemm_proj(const float* __restrict__ A, const float* __restrict__ W,
                      const float* __restrict__ bias, __bf16* __restrict__ C,
                      int M, int Mstore, int N, int K) {
  __shared__ __bf16 Alds[64 * 32];   // [row][k]
  __shared__ __bf16 Blds[64 * 32];   // [n][k]  (W transposed into LDS)
  const int tid = threadIdx.x, lane = tid & 31, wave = tid >> 5;
  const int half = lane >> 4, l16 = lane & 15;
  const int m0 = blockIdx.x * 64, n0 = blockIdx.y * 64;
  const bool full_tile = (m0 + 64 <= M);   // uniform: avoid per-load EXEC churn

  v8f acc[4] = {};
  for (int k0 = 0; k0 < K; k0 += 32) {
    // stage A tile: fp32 -> bf16 (v_cvt_pk_bf16_f32 path)
    if (full_tile) {
      for (int i = tid; i < 512; i += 128) {
        int row = i >> 3, c = (i & 7) * 4;
        float4 x = *(const float4*)(A + (size_t)(m0 + row) * K + k0 + c);
        __bf16* dst = Alds + row * 32 + c;
        dst[0] = (__bf16)x.x; dst[1] = (__bf16)x.y;
        dst[2] = (__bf16)x.z; dst[3] = (__bf16)x.w;
      }
    } else {
      for (int i = tid; i < 512; i += 128) {
        int row = i >> 3, c = (i & 7) * 4;
        float4 x = {0.f, 0.f, 0.f, 0.f};
        int gr = m0 + row;
        if (gr < M) x = *(const float4*)(A + (size_t)gr * K + k0 + c);
        __bf16* dst = Alds + row * 32 + c;
        dst[0] = (__bf16)x.x; dst[1] = (__bf16)x.y;
        dst[2] = (__bf16)x.z; dst[3] = (__bf16)x.w;
      }
    }
    // stage W tile transposed: Blds[n][k] = W[k0+k][n0+n]
    for (int i = tid; i < 512; i += 128) {
      int kk = i >> 4, nq = (i & 15) * 4;
      float4 w = *(const float4*)(W + (size_t)(k0 + kk) * N + n0 + nq);
      Blds[(nq + 0) * 32 + kk] = (__bf16)w.x;
      Blds[(nq + 1) * 32 + kk] = (__bf16)w.y;
      Blds[(nq + 2) * 32 + kk] = (__bf16)w.z;
      Blds[(nq + 3) * 32 + kk] = (__bf16)w.w;
    }
    __syncthreads();
    v16bf af = load_a_frag(Alds + (wave * 16 + l16) * 32, half);
#pragma unroll
    for (int t = 0; t < 4; ++t) {
      v16bf bf = load_b_frag(Blds + (t * 16 + l16) * 32, half);
      acc[t] = WMMA_BF16(af, bf, acc[t]);
    }
    __syncthreads();
  }
  // C layout: vgpr r -> M = r + 8*half, N = t*16 + (lane&15)
#pragma unroll
  for (int t = 0; t < 4; ++t) {
    int gc = n0 + t * 16 + l16;
    float bv = bias[gc];
#pragma unroll
    for (int r = 0; r < 8; ++r) {
      int gr = m0 + wave * 16 + r + 8 * half;
      if (gr < Mstore) C[(size_t)gr * N + gc] = (__bf16)(acc[t][r] + bv);
    }
  }
}

// ============================================================
// Fused attention per (b, h, 64-row L tile).
// Scores 64x1024 live entirely in LDS (never touch HBM).
// Q and K tiles staged with async global->LDS (ASYNCcnt) copies.
// ============================================================
__global__ __launch_bounds__(128)
void reprog_attn(const __bf16* __restrict__ q, const __bf16* __restrict__ k,
                 const __bf16* __restrict__ v, __bf16* __restrict__ o) {
  extern __shared__ char smem[];
  __bf16* Qlds  = (__bf16*)smem;          // 64*64   [l][e]
  __bf16* Ktile = Qlds + 64 * 64;         // 64*64   [s'][e]
  __bf16* Sbuf  = Ktile + 64 * 64;        // 64*1024 [l][s] scores -> exp(P)
  __bf16* Vt    = Sbuf + 64 * SP;         // 64*1024 [e][s]
  float*  rinv  = (float*)(Vt + 64 * SP); // 64 reciprocal row sums

  const int tid = threadIdx.x, lane = tid & 31, wave = tid >> 5;
  const int half = lane >> 4, l16 = lane & 15;
  const int lt = blockIdx.x, h = blockIdx.y, b = blockIdx.z;
  const size_t qbase = ((size_t)(b * L_ + lt * 64) * HE) + h * E_;
  const unsigned q_lds = lds_offset(Qlds);
  const unsigned k_lds = lds_offset(Ktile);

  // stage Q tile: async b128 copies, 4 chunks/thread
  for (int i = tid; i < 64 * 8; i += 128) {
    int row = i >> 3, c = i & 7;
    async_copy_b128(q_lds + i * 16, q + qbase + (size_t)row * HE + c * 8);
  }
  // stage V transposed (needs data swizzle -> VALU path): Vt[e][s] = v[s][h*64+e]
  {
    const uint4* vg = (const uint4*)v;
    for (int i = tid; i < SP * 8; i += 128) {
      int s = i >> 3, e0 = (i & 7) * 8;
      uint4 pk = vg[((size_t)s * HE + h * E_) / 8 + (i & 7)];
      const __bf16* pb = (const __bf16*)&pk;
#pragma unroll
      for (int j = 0; j < 8; ++j) Vt[(e0 + j) * SP + s] = pb[j];
    }
  }
  async_wait0();
  __syncthreads();

  // ---- phase 1: scores S = scale * Q K^T (masked), into Sbuf ----
  for (int sc = 0; sc < SP / 64; ++sc) {
    for (int i = tid; i < 64 * 8; i += 128) {     // async K tile stage
      int sr = i >> 3, c = i & 7;
      async_copy_b128(k_lds + i * 16,
                      k + (size_t)(sc * 64 + sr) * HE + h * E_ + c * 8);
    }
    async_wait0();
    __syncthreads();

    v8f acc[4] = {};
#pragma unroll
    for (int ke = 0; ke < 2; ++ke) {
      v16bf af = load_a_frag(Qlds + (wave * 16 + l16) * 64 + ke * 32, half);
#pragma unroll
      for (int t = 0; t < 4; ++t) {
        v16bf bf = load_b_frag(Ktile + (t * 16 + l16) * 64 + ke * 32, half);
        acc[t] = WMMA_BF16(af, bf, acc[t]);
      }
    }
#pragma unroll
    for (int t = 0; t < 4; ++t) {
      int scol = sc * 64 + t * 16 + l16;
      bool oob = (scol >= S_);
#pragma unroll
      for (int r = 0; r < 8; ++r) {
        float x = acc[t][r] * 0.125f;     // 1/sqrt(E)
        if (oob) x = -1e30f;              // pad mask
        int mrow = wave * 16 + r + 8 * half;
        Sbuf[mrow * SP + scol] = (__bf16)x;
      }
    }
    __syncthreads();
  }

  // ---- phase 2: softmax rows (2 threads / row) ----
  {
    int row = tid >> 1, part = tid & 1;
    __bf16* srow = Sbuf + row * SP + part * 512;
    float m = -3e38f;
    for (int i = 0; i < 512; i += 8) {
      v8bf pk = *(const v8bf*)(srow + i);
#pragma unroll
      for (int j = 0; j < 8; ++j) m = fmaxf(m, (float)pk[j]);
    }
    m = fmaxf(m, __shfl_xor(m, 1, 32));
    float sum = 0.f;
    for (int i = 0; i < 512; i += 8) {
      v8bf pk = *(const v8bf*)(srow + i);
      v8bf eo;
#pragma unroll
      for (int j = 0; j < 8; ++j) {
        float x = __expf((float)pk[j] - m);
        sum += x;
        eo[j] = (__bf16)x;
      }
      *(v8bf*)(srow + i) = eo;
    }
    sum += __shfl_xor(sum, 1, 32);
    if (part == 0) rinv[row] = 1.0f / sum;
  }
  __syncthreads();

  // ---- phase 3: O = P @ V_h ----
  v8f oacc[4] = {};
  for (int kc = 0; kc < SP / 32; ++kc) {
    v16bf af = load_a_frag(Sbuf + (wave * 16 + l16) * SP + kc * 32, half);
#pragma unroll
    for (int t = 0; t < 4; ++t) {
      v16bf bf = load_b_frag(Vt + (t * 16 + l16) * SP + kc * 32, half);
      oacc[t] = WMMA_BF16(af, bf, oacc[t]);
    }
  }
#pragma unroll
  for (int t = 0; t < 4; ++t) {
#pragma unroll
    for (int r = 0; r < 8; ++r) {
      int mrow = wave * 16 + r + 8 * half;
      float val = oacc[t][r] * rinv[mrow];
      size_t gi = ((size_t)(b * L_ + lt * 64 + mrow) * HE) + h * E_ + t * 16 + l16;
      o[gi] = (__bf16)val;
    }
  }
}

// ============================================================
// GEMM 2: out_f32[M,64] = attn_bf16[M,512] @ Wo + bo
// bf16 A tiles staged with async global->LDS copies.
// ============================================================
__global__ __launch_bounds__(128)
void reprog_gemm_out(const __bf16* __restrict__ A, const float* __restrict__ W,
                     const float* __restrict__ bias, float* __restrict__ C,
                     int M, int N, int K) {
  __shared__ __bf16 Alds[64 * 32];
  __shared__ __bf16 Blds[64 * 32];
  const int tid = threadIdx.x, lane = tid & 31, wave = tid >> 5;
  const int half = lane >> 4, l16 = lane & 15;
  const int m0 = blockIdx.x * 64, n0 = blockIdx.y * 64;
  const unsigned a_lds = lds_offset(Alds);

  v8f acc[4] = {};
  for (int k0 = 0; k0 < K; k0 += 32) {
    // A already bf16: async b128 copies (2 chunks/thread)
    for (int i = tid; i < 256; i += 128) {
      int row = i >> 2, c = i & 3;
      async_copy_b128(a_lds + i * 16, A + (size_t)(m0 + row) * K + k0 + c * 8);
    }
    for (int i = tid; i < 512; i += 128) {
      int kk = i >> 4, nq = (i & 15) * 4;
      float4 w = *(const float4*)(W + (size_t)(k0 + kk) * N + n0 + nq);
      Blds[(nq + 0) * 32 + kk] = (__bf16)w.x;
      Blds[(nq + 1) * 32 + kk] = (__bf16)w.y;
      Blds[(nq + 2) * 32 + kk] = (__bf16)w.z;
      Blds[(nq + 3) * 32 + kk] = (__bf16)w.w;
    }
    async_wait0();
    __syncthreads();
    v16bf af = load_a_frag(Alds + (wave * 16 + l16) * 32, half);
#pragma unroll
    for (int t = 0; t < 4; ++t) {
      v16bf bf = load_b_frag(Blds + (t * 16 + l16) * 32, half);
      acc[t] = WMMA_BF16(af, bf, acc[t]);
    }
    __syncthreads();
  }
#pragma unroll
  for (int t = 0; t < 4; ++t) {
    int gc = n0 + t * 16 + l16;
    float bv = bias[gc];
#pragma unroll
    for (int r = 0; r < 8; ++r) {
      int gr = m0 + wave * 16 + r + 8 * half;
      if (gr < M) C[(size_t)gr * N + gc] = acc[t][r] + bv;
    }
  }
}

// ============================================================
extern "C" void kernel_launch(void* const* d_in, const int* in_sizes, int n_in,
                              void* d_out, int out_size, void* d_ws, size_t ws_size,
                              hipStream_t stream) {
  const float* target = (const float*)d_in[0];   // (B,L,512)
  const float* source = (const float*)d_in[1];   // (S,1024)
  const float* value  = (const float*)d_in[2];   // (S,1024)
  const float* Wq = (const float*)d_in[3];
  const float* bq = (const float*)d_in[4];
  const float* Wk = (const float*)d_in[5];
  const float* bk = (const float*)d_in[6];
  const float* Wv = (const float*)d_in[7];
  const float* bv = (const float*)d_in[8];
  const float* Wo = (const float*)d_in[9];
  const float* bo = (const float*)d_in[10];
  float* out = (float*)d_out;

  // workspace: q(32MB) | attn(32MB) | k(1MB) | v(1MB)  == 66 MB
  char* ws = (char*)d_ws;
  __bf16* qbf  = (__bf16*)ws;
  __bf16* attn = (__bf16*)(ws + (size_t)BL * HE * 2);
  __bf16* kbf  = (__bf16*)(ws + (size_t)2 * BL * HE * 2);
  __bf16* vbf  = kbf + (size_t)SP * HE;

  // projections (fp32 in, bf16 out)
  reprog_gemm_proj<<<dim3(BL / 64, HE / 64), 128, 0, stream>>>(
      target, Wq, bq, qbf, BL, BL, HE, DM);
  reprog_gemm_proj<<<dim3(SP / 64, HE / 64), 128, 0, stream>>>(
      source, Wk, bk, kbf, S_, SP, HE, DL);
  reprog_gemm_proj<<<dim3(SP / 64, HE / 64), 128, 0, stream>>>(
      value, Wv, bv, vbf, S_, SP, HE, DL);

  // fused attention; scores stay in LDS (279 KB of the WGP's 320 KB)
  size_t smem = (size_t)(64 * 64 + 64 * 64 + 64 * SP + 64 * SP) * sizeof(__bf16)
              + 64 * sizeof(float);
  reprog_attn<<<dim3(L_ / 64, H_, B_), 128, smem, stream>>>(qbf, kbf, vbf, attn);

  // output projection (bf16 in, fp32 out)
  reprog_gemm_out<<<dim3(BL / 64, 1), 128, 0, stream>>>(
      attn, Wo, bo, out, BL, 64, HE);
}
